// Attention_82815559401482
// MI455X (gfx1250) — matrix operand order repaired and checked by reading, not compiled, not run on hardware
//
#include <hip/hip_runtime.h>
#include <hip/hip_bf16.h>

#define BB     16
#define NN     784
#define DIMC   512
#define HH     8
#define KDIM   32
#define DD     128
#define DHH    1024
#define EPSF   1e-5f
#define SCALEF 0.17677669529663687f
#define NEGBIG (-1.0e30f)

typedef __attribute__((ext_vector_type(16))) __bf16 v16bf;
typedef __attribute__((ext_vector_type(8)))  __bf16 v8bf;
typedef __attribute__((ext_vector_type(4)))  __bf16 v4bf;
typedef __attribute__((ext_vector_type(8)))  float  v8f;
typedef __attribute__((ext_vector_type(2)))  int    vi2;
typedef __attribute__((ext_vector_type(4)))  int    vi4;

// --- CDNA5 async global->LDS copies (ASYNCcnt), with synchronous fallback ---
#if defined(__has_builtin)
#if __has_builtin(__builtin_amdgcn_global_load_async_to_lds_b64) && \
    __has_builtin(__builtin_amdgcn_global_load_async_to_lds_b128)
#define HAVE_ASYNC_LDS 1
#endif
#endif
#ifndef HAVE_ASYNC_LDS
#define HAVE_ASYNC_LDS 0
#endif

#if HAVE_ASYNC_LDS && __has_builtin(__builtin_amdgcn_s_wait_asynccnt)
#define WAIT_ASYNC() __builtin_amdgcn_s_wait_asynccnt(0)
#elif HAVE_ASYNC_LDS
#define WAIT_ASYNC() asm volatile("s_wait_asynccnt 0" ::: "memory")
#else
#define WAIT_ASYNC() ((void)0)
#endif

// Builtin parameter types (from hipcc diagnostics): typed vector pointers,
// global side in AS1, LDS side in AS3.
typedef __attribute__((address_space(1))) vi2 g_vi2;
typedef __attribute__((address_space(3))) vi2 l_vi2;
typedef __attribute__((address_space(1))) vi4 g_vi4;
typedef __attribute__((address_space(3))) vi4 l_vi4;

__device__ __forceinline__ void async_b64(const void* g, void* l) {
#if HAVE_ASYNC_LDS
  __builtin_amdgcn_global_load_async_to_lds_b64((g_vi2*)g, (l_vi2*)l, 0, 0);
#else
  *(v4bf*)l = *(const v4bf*)g;  // 8 bytes
#endif
}
__device__ __forceinline__ void async_b128(const void* g, void* l) {
#if HAVE_ASYNC_LDS
  __builtin_amdgcn_global_load_async_to_lds_b128((g_vi4*)g, (l_vi4*)l, 0, 0);
#else
  *(float4*)l = *(const float4*)g;  // 16 bytes
#endif
}

__device__ __forceinline__ v8f wmma_bf16(v16bf a, v16bf b, v8f c) {
  return __builtin_amdgcn_wmma_f32_16x16x32_bf16(false, a, false, b, (short)0, c,
                                                 false, false);
}

__device__ __forceinline__ v16bf cat16(v8bf lo, v8bf hi) {
  return __builtin_shufflevector(lo, hi, 0, 1, 2, 3, 4, 5, 6, 7, 8, 9, 10, 11,
                                 12, 13, 14, 15);
}

// 8 contiguous f32 (global or LDS) -> v8bf via two b128 loads.
__device__ __forceinline__ v8bf cvt8f(const float* __restrict__ p) {
  const float4 f0 = *(const float4*)p;
  const float4 f1 = *(const float4*)(p + 4);
  v8bf r;
  r[0] = (__bf16)f0.x; r[1] = (__bf16)f0.y; r[2] = (__bf16)f0.z; r[3] = (__bf16)f0.w;
  r[4] = (__bf16)f1.x; r[5] = (__bf16)f1.y; r[6] = (__bf16)f1.z; r[7] = (__bf16)f1.w;
  return r;
}

// Per-wave LDS fence: wait for our ds stores, block compiler reordering.
__device__ __forceinline__ void wave_lds_fence() {
  asm volatile("s_wait_dscnt 0" ::: "memory");
  __builtin_amdgcn_wave_barrier();
}

// ---------------------------------------------------------------------------
// Kernel 0: expand attention bias to a dense bf16 (H, N, N) table.
// ---------------------------------------------------------------------------
__global__ __launch_bounds__(256) void bias_expand_kernel(
    const float* __restrict__ ab, const int* __restrict__ bias_idx,
    __bf16* __restrict__ btab) {
  const size_t e = ((size_t)blockIdx.x * 256 + threadIdx.x) * 4;
  const int h  = (int)(e / ((size_t)NN * NN));
  const int rc = (int)(e % ((size_t)NN * NN));
  const int4 i4 = *(const int4*)&bias_idx[rc];
  const float* __restrict__ abh = ab + h * NN;
  v4bf o;
  o[0] = (__bf16)abh[i4.x];
  o[1] = (__bf16)abh[i4.y];
  o[2] = (__bf16)abh[i4.z];
  o[3] = (__bf16)abh[i4.w];
  *(v4bf*)&btab[e] = o;
}

// ---------------------------------------------------------------------------
// Kernel 1: QKV GEMM (x @ w_qkv^T) + BN, scatter to bf16 q/k/v (B,H,N,*).
// Block = 8 waves = 16 rows x 128 cols. Raw f32 A tile async-staged to LDS,
// double buffered; f32->bf16 conversion done at fragment build.
// ---------------------------------------------------------------------------
__global__ __launch_bounds__(256) void qkv_kernel(
    const float* __restrict__ x, const float* __restrict__ w,
    const float* __restrict__ g, const float* __restrict__ be,
    const float* __restrict__ mu, const float* __restrict__ va,
    __bf16* __restrict__ qb, __bf16* __restrict__ kb, __bf16* __restrict__ vb) {
  __shared__ float at[2][16][32];
  const int wave = threadIdx.x >> 5, lane = threadIdx.x & 31;
  const int nl = lane & 15, hi = lane >> 4;
  const int t = threadIdx.x;
  const int row0 = blockIdx.x * 16;
  const int col0 = blockIdx.y * 128 + wave * 16;

  auto stageA = [&](int kbs, int buf) {
    if (t < 128) {  // 16x32 f32 = 128 threads x b128
      const int r = t >> 3, kc = (t & 7) * 4;
      async_b128(x + (size_t)(row0 + r) * DIMC + kbs + kc, &at[buf][r][kc]);
    }
  };

  v8f acc = {};
  const float* __restrict__ wr = w + (size_t)(col0 + nl) * DIMC;
  stageA(0, 0);
  for (int step = 0; step < DIMC / 32; ++step) {
    const int kbs = step * 32, cur = step & 1;
    WAIT_ASYNC();
    __syncthreads();
    if (kbs + 32 < DIMC) stageA(kbs + 32, cur ^ 1);
    const v16bf a = cat16(cvt8f(&at[cur][nl][hi * 16]),
                          cvt8f(&at[cur][nl][hi * 16 + 8]));
    const v16bf b = cat16(cvt8f(wr + kbs + hi * 16),
                          cvt8f(wr + kbs + hi * 16 + 8));
    acc = wmma_bf16(a, b, acc);
  }

  const int c  = col0 + nl;
  const float sc = g[c] * rsqrtf(va[c] + EPSF);
  const float bt = be[c] - mu[c] * sc;
  const int h = c / 192, r = c % 192;
#pragma unroll
  for (int rr = 0; rr < 8; ++rr) {
    const int grow = row0 + rr + hi * 8;  // global row in [0, B*N)
    const int bidx = grow / NN, n = grow % NN;
    const float val = acc[rr] * sc + bt;
    const size_t hn = ((size_t)bidx * HH + h) * NN + n;
    if (r < KDIM)
      qb[hn * KDIM + r] = (__bf16)val;
    else if (r < 2 * KDIM)
      kb[hn * KDIM + (r - KDIM)] = (__bf16)val;
    else
      vb[hn * DD + (r - 2 * KDIM)] = (__bf16)val;
  }
}

// ---------------------------------------------------------------------------
// Kernel 2: flash-attention per (b,h). Block = 8 waves, one 16-row tile each.
// K/V tiles async-staged to LDS (double buffered) and overlapped with the
// 2 score WMMAs + online softmax + 8 P*V WMMAs of the previous tile.
// ---------------------------------------------------------------------------
__global__ __launch_bounds__(256) void attn_kernel(
    const __bf16* __restrict__ qb, const __bf16* __restrict__ kb,
    const __bf16* __restrict__ vb, const __bf16* __restrict__ btab,
    __bf16* __restrict__ actb) {
  __shared__ __bf16 kt[2][32][KDIM];   // [buf][local col][kd]
  __shared__ __bf16 vt[2][32][DD];     // [buf][local row kk][d]
  __shared__ __bf16 pt[8][16][32];     // per-wave P tile [m][kk]

  const int wave = threadIdx.x >> 5, lane = threadIdx.x & 31;
  const int nl = lane & 15, hi = lane >> 4;
  const int bh = blockIdx.x;
  const int b = bh / HH, h = bh % HH;
  const int rowtile = blockIdx.y * 8 + wave;
  const bool valid = rowtile < 49;     // 784/16 = 49 row tiles
  const int row0 = rowtile * 16;
  const size_t bhbase = (size_t)bh * NN;
  const __bf16* __restrict__ bias = btab + (size_t)h * NN * NN;
  const int t = threadIdx.x;

  auto stageKV = [&](int ct, int buf) {
    const int c0 = ct * 32;
    {  // K tile: 32 cols x 32 kd, one b64 per thread
      const int ncl = t >> 3, kdq = (t & 7) * 4;
      int col = c0 + ncl;
      if (col >= NN) col = NN - 1;
      async_b64(&kb[(bhbase + col) * KDIM + kdq], &kt[buf][ncl][kdq]);
    }
#pragma unroll
    for (int j = 0; j < 2; ++j) {  // V tile: 32 rows x 128 d, two b128
      const int q = j * 256 + t;
      const int kk = q >> 4, dq = (q & 15) * 8;
      int col = c0 + kk;
      if (col >= NN) col = NN - 1;
      async_b128(&vb[(bhbase + col) * DD + dq], &vt[buf][kk][dq]);
    }
  };

  // Q fragment (A layout, 16x32 bf16): two v8bf global loads.
  v16bf aq = {};
  if (valid) {
    const __bf16* __restrict__ qr = qb + (bhbase + row0 + nl) * KDIM;
    aq = cat16(*(const v8bf*)(qr + hi * 16), *(const v8bf*)(qr + hi * 16 + 8));
  }

  v8f acc[8];
  const v8f vzero = {};
#pragma unroll
  for (int i = 0; i < 8; ++i) acc[i] = vzero;
  float mrow[8], lrow[8];
#pragma unroll
  for (int i = 0; i < 8; ++i) { mrow[i] = NEGBIG; lrow[i] = 0.f; }

  stageKV(0, 0);
  for (int ct = 0; ct < 25; ++ct) {
    const int c0 = ct * 32, cur = ct & 1;
    WAIT_ASYNC();
    __syncthreads();
    if (ct + 1 < 25) stageKV(ct + 1, cur ^ 1);
    if (!valid) continue;

    // Scores: S = Q K^T for two 16-col halves (KD=32 = one WMMA K-step).
    const v16bf bk0 = cat16(*(const v8bf*)&kt[cur][nl][hi * 16],
                            *(const v8bf*)&kt[cur][nl][hi * 16 + 8]);
    const v16bf bk1 = cat16(*(const v8bf*)&kt[cur][nl + 16][hi * 16],
                            *(const v8bf*)&kt[cur][nl + 16][hi * 16 + 8]);
    v8f s0 = wmma_bf16(aq, bk0, vzero);
    v8f s1 = wmma_bf16(aq, bk1, vzero);

    const int colA = c0 + nl, colB = c0 + 16 + nl;
#pragma unroll
    for (int rr = 0; rr < 8; ++rr) {
      const int row = row0 + rr + 8 * hi;
      float sv0 = NEGBIG, sv1 = NEGBIG;
      if (colA < NN) sv0 = s0[rr] * SCALEF + (float)bias[(size_t)row * NN + colA];
      if (colB < NN) sv1 = s1[rr] * SCALEF + (float)bias[(size_t)row * NN + colB];
      float tm = fmaxf(sv0, sv1);
#pragma unroll
      for (int off = 1; off < 16; off <<= 1) tm = fmaxf(tm, __shfl_xor(tm, off, 32));
      const float mnew = fmaxf(mrow[rr], tm);
      const float corr = __expf(mrow[rr] - mnew);
      mrow[rr] = mnew;
      const float p0 = __expf(sv0 - mnew);
      const float p1 = __expf(sv1 - mnew);
      lrow[rr] = lrow[rr] * corr + p0 + p1;
#pragma unroll
      for (int n8 = 0; n8 < 8; ++n8) acc[n8][rr] *= corr;
      pt[wave][rr + 8 * hi][nl]      = (__bf16)p0;
      pt[wave][rr + 8 * hi][16 + nl] = (__bf16)p1;
    }
    wave_lds_fence();

    // Re-layout P (C layout -> A layout) via LDS, then P @ V.
    const v16bf ap = cat16(*(const v8bf*)&pt[wave][nl][hi * 16],
                           *(const v8bf*)&pt[wave][nl][hi * 16 + 8]);
#pragma unroll
    for (int n8 = 0; n8 < 8; ++n8) {
      v16bf bv;
      const int d = n8 * 16 + nl;
#pragma unroll
      for (int i = 0; i < 16; ++i) bv[i] = vt[cur][hi * 16 + i][d];
      acc[n8] = wmma_bf16(ap, bv, acc[n8]);
    }
  }

  if (!valid) return;
#pragma unroll
  for (int rr = 0; rr < 8; ++rr) {
    float ls = lrow[rr];
#pragma unroll
    for (int off = 1; off < 16; off <<= 1) ls += __shfl_xor(ls, off, 32);
    const float inv = 1.f / ls;
    const int row = row0 + rr + 8 * hi;
    __bf16* __restrict__ orow = actb + ((size_t)b * NN + row) * DHH + h * DD;
#pragma unroll
    for (int n8 = 0; n8 < 8; ++n8) {
      const float v = acc[n8][rr] * inv;
      const float hs = v * fminf(fmaxf(v + 3.f, 0.f), 6.f) * (1.f / 6.f);
      orow[n8 * 16 + nl] = (__bf16)hs;
    }
  }
}

// ---------------------------------------------------------------------------
// Kernel 3: projection GEMM (act @ w_proj^T) + BN, f32 output (B, N, DIM).
// bf16 A tile async-staged to LDS, double buffered.
// ---------------------------------------------------------------------------
__global__ __launch_bounds__(256) void proj_kernel(
    const __bf16* __restrict__ act, const float* __restrict__ w,
    const float* __restrict__ g, const float* __restrict__ be,
    const float* __restrict__ mu, const float* __restrict__ va,
    float* __restrict__ out) {
  __shared__ __bf16 at[2][16][32];
  const int wave = threadIdx.x >> 5, lane = threadIdx.x & 31;
  const int nl = lane & 15, hi = lane >> 4;
  const int t = threadIdx.x;
  const int row0 = blockIdx.x * 16;
  const int col0 = blockIdx.y * 128 + wave * 16;

  auto stageA = [&](int kbs, int buf) {
    if (t < 64) {  // 512 bf16 = 64 threads x b128
      const int r = t >> 2, kc = (t & 3) * 8;
      async_b128(&act[(size_t)(row0 + r) * DHH + kbs + kc], &at[buf][r][kc]);
    }
  };

  v8f acc = {};
  const float* __restrict__ wr = w + (size_t)(col0 + nl) * DHH;
  stageA(0, 0);
  for (int step = 0; step < DHH / 32; ++step) {
    const int kbs = step * 32, cur = step & 1;
    WAIT_ASYNC();
    __syncthreads();
    if (kbs + 32 < DHH) stageA(kbs + 32, cur ^ 1);
    const v16bf a = cat16(*(const v8bf*)&at[cur][nl][hi * 16],
                          *(const v8bf*)&at[cur][nl][hi * 16 + 8]);
    const v16bf b = cat16(cvt8f(wr + kbs + hi * 16),
                          cvt8f(wr + kbs + hi * 16 + 8));
    acc = wmma_bf16(a, b, acc);
  }

  const int c = col0 + nl;
  const float sc = g[c] * rsqrtf(va[c] + EPSF);
  const float bt = be[c] - mu[c] * sc;
#pragma unroll
  for (int rr = 0; rr < 8; ++rr) {
    const int grow = row0 + rr + hi * 8;
    out[(size_t)grow * DIMC + c] = acc[rr] * sc + bt;
  }
}

// ---------------------------------------------------------------------------
extern "C" void kernel_launch(void* const* d_in, const int* in_sizes, int n_in,
                              void* d_out, int out_size, void* d_ws, size_t ws_size,
                              hipStream_t stream) {
  (void)in_sizes; (void)n_in; (void)out_size; (void)ws_size;
  const float* x      = (const float*)d_in[0];
  const float* w_qkv  = (const float*)d_in[1];
  const float* qkv_g  = (const float*)d_in[2];
  const float* qkv_b  = (const float*)d_in[3];
  const float* qkv_m  = (const float*)d_in[4];
  const float* qkv_v  = (const float*)d_in[5];
  const float* ab     = (const float*)d_in[6];
  const float* w_proj = (const float*)d_in[7];
  const float* proj_g = (const float*)d_in[8];
  const float* proj_b = (const float*)d_in[9];
  const float* proj_m = (const float*)d_in[10];
  const float* proj_v = (const float*)d_in[11];
  const int*   bidx   = (const int*)d_in[12];
  float* out = (float*)d_out;

  // Workspace layout (bf16): q, k, v, act, bias table. ~74 MB total.
  __bf16* ws = (__bf16*)d_ws;
  const size_t qn = (size_t)BB * HH * NN * KDIM;   // 3,211,264
  const size_t vn = (size_t)BB * HH * NN * DD;     // 12,845,056
  const size_t an = (size_t)BB * NN * DHH;         // 12,845,056
  __bf16* qbuf = ws;
  __bf16* kbuf = qbuf + qn;
  __bf16* vbuf = kbuf + qn;
  __bf16* actb = vbuf + vn;
  __bf16* btab = actb + an;                        // H*N*N = 4,917,248

  bias_expand_kernel<<<dim3(4802), 256, 0, stream>>>(ab, bidx, btab);
  qkv_kernel<<<dim3(784, 12), 256, 0, stream>>>(
      x, w_qkv, qkv_g, qkv_b, qkv_m, qkv_v, qbuf, kbuf, vbuf);
  attn_kernel<<<dim3(128, 7), 256, 0, stream>>>(
      qbuf, kbuf, vbuf, btab, actb);
  proj_kernel<<<dim3(784, 4), 256, 0, stream>>>(
      actb, w_proj, proj_g, proj_b, proj_m, proj_v, out);
}
